// PathCausalSelfAttention_53377853555169
// MI455X (gfx1250) — matrix operand order repaired and checked by reading, not compiled, not run on hardware
//
#include <hip/hip_runtime.h>
#include <math.h>

// ---------------------------------------------------------------------------
// PathCausalSelfAttention for MI455X (gfx1250, wave32, WMMA f16 16x16x32)
//   B=2, L=2048, D=1024, H=16, HD=64
//   GEMMs register-blocked 32x32 per wave (2x2 WMMA tiles, 2x fragment reuse)
// ---------------------------------------------------------------------------

typedef __attribute__((ext_vector_type(16))) _Float16 v16h;
typedef __attribute__((ext_vector_type(8)))  float    v8f;

constexpr int Bn  = 2;
constexpr int Ln  = 2048;
constexpr int Dn  = 1024;
constexpr int Hn  = 16;
constexpr int HDn = 64;
constexpr int M1  = Bn * Ln;   // 4096 rows
constexpr int N1  = 3 * Dn;    // 3072 qkv cols
constexpr float SCALE = 0.125f;  // 1/sqrt(64)

// f16 workspace layout (element offsets)
constexpr size_t XB_OFF    = 0;                                   // [M1][Dn]
constexpr size_t WQKVT_OFF = XB_OFF    + (size_t)M1 * Dn;         // [N1][Dn] (transposed)
constexpr size_t WOUTT_OFF = WQKVT_OFF + (size_t)N1 * Dn;         // [Dn][Dn] (transposed)
constexpr size_t GB_OFF    = WOUTT_OFF + (size_t)Dn * Dn;         // [B,H,L,HD]
constexpr size_t QB_OFF    = GB_OFF    + (size_t)Bn * Hn * Ln * HDn;
constexpr size_t KB_OFF    = QB_OFF    + (size_t)Bn * Hn * Ln * HDn;
constexpr size_t VTB_OFF   = KB_OFF    + (size_t)Bn * Hn * Ln * HDn; // [B,H,HD,L]
constexpr size_t YB_OFF    = VTB_OFF   + (size_t)Bn * Hn * Ln * HDn; // [B,L,D]

union FragH { v16h v; uint4 q[2]; };
union AccF  { v8f  v; float f[8]; };

// A-matrix fragment (16x32 f16, row-major source, `stride` elems per row).
// lane<16: row=lane, K in {0..7, 16..23}; lane>=16: row=lane-16, K in {8..15, 24..31}
static __device__ inline v16h load_fragA(const _Float16* tile, int stride, int lane) {
    const int half = lane >> 4;
    const int r    = lane & 15;
    const _Float16* p = tile + (size_t)r * stride + half * 8;
    FragH f;
    f.q[0] = *reinterpret_cast<const uint4*>(p);
    f.q[1] = *reinterpret_cast<const uint4*>(p + 16);
    return f.v;
}

// B-matrix fragment (32x16 f16) from K-major (transposed) storage Bt[n][k].
// lane<16: n=lane, K in {0..7, 16..23}; lane>=16: n=lane-16, K in {8..15, 24..31}
static __device__ inline v16h load_fragB(const _Float16* tileT, int stride, int lane) {
    const int half = lane >> 4;
    const int n    = lane & 15;
    const _Float16* p = tileT + (size_t)n * stride + half * 8;
    FragH f;
    f.q[0] = *reinterpret_cast<const uint4*>(p);
    f.q[1] = *reinterpret_cast<const uint4*>(p + 16);
    return f.v;
}

static __device__ inline v8f wmma16(v16h a, v16h b, v8f c) {
    return __builtin_amdgcn_wmma_f32_16x16x32_f16(
        /*neg_a=*/false, a, /*neg_b=*/false, b,
        /*c_mod=*/(short)0, c, /*reuse_a=*/false, /*reuse_b=*/false);
}

// ---------------------------------------------------------------------------
// 1) fp32 -> f16 conversions (x, g->head layout, W_qkv^T, W_out^T)
// ---------------------------------------------------------------------------
__global__ __launch_bounds__(256) void convert_kernel(
    const float* __restrict__ x, const float* __restrict__ g,
    const float* __restrict__ wqkv, const float* __restrict__ wout,
    _Float16* __restrict__ xb, _Float16* __restrict__ gb,
    _Float16* __restrict__ wqkvt, _Float16* __restrict__ woutt) {
    const int idx = blockIdx.x * blockDim.x + threadIdx.x;
    if (idx < M1 * Dn) {
        xb[idx] = (_Float16)x[idx];
        const int b   = idx / (Ln * Dn);
        const int rem = idx % (Ln * Dn);
        const int l   = rem / Dn;
        const int d   = rem % Dn;
        const int h   = d / HDn;
        const int hd  = d % HDn;
        gb[(((size_t)(b * Hn + h)) * Ln + l) * HDn + hd] = (_Float16)g[idx];
    }
    if (idx < Dn * N1) {
        const int k = idx / N1;
        const int n = idx % N1;
        wqkvt[(size_t)n * Dn + k] = (_Float16)wqkv[idx];
    }
    if (idx < Dn * Dn) {
        const int k = idx / Dn;
        const int n = idx % Dn;
        woutt[(size_t)n * Dn + k] = (_Float16)wout[idx];
    }
}

// ---------------------------------------------------------------------------
// 2) QKV projection: Xb[4096x1024] @ WqkvT -> scatter Q/K [B,H,L,HD], Vt [B,H,HD,L]
//    block = 4 waves; each wave: 32x32 C macro-tile (2x2 WMMA tiles)
// ---------------------------------------------------------------------------
__global__ __launch_bounds__(128) void qkv_gemm_kernel(
    const _Float16* __restrict__ A, const _Float16* __restrict__ Bt,
    _Float16* __restrict__ Qb, _Float16* __restrict__ Kb, _Float16* __restrict__ Vtb) {
    const int lane = threadIdx.x & 31;
    const int wave = threadIdx.x >> 5;
    const int m0   = blockIdx.y * 32;
    const int n0   = blockIdx.x * 128 + wave * 32;

    AccF acc[2][2];
#pragma unroll
    for (int mi = 0; mi < 2; ++mi)
#pragma unroll
        for (int ni = 0; ni < 2; ++ni)
#pragma unroll
            for (int i = 0; i < 8; ++i) acc[mi][ni].f[i] = 0.0f;

    const _Float16* arow = A  + (size_t)m0 * Dn;
    const _Float16* brow = Bt + (size_t)n0 * Dn;
    for (int k0 = 0; k0 < Dn; k0 += 32) {
        v16h a0 = load_fragA(arow + k0,                    Dn, lane);
        v16h a1 = load_fragA(arow + (size_t)16 * Dn + k0,  Dn, lane);
        v16h b0 = load_fragB(brow + k0,                    Dn, lane);
        v16h b1 = load_fragB(brow + (size_t)16 * Dn + k0,  Dn, lane);
        acc[0][0].v = wmma16(a0, b0, acc[0][0].v);
        acc[0][1].v = wmma16(a0, b1, acc[0][1].v);
        acc[1][0].v = wmma16(a1, b0, acc[1][0].v);
        acc[1][1].v = wmma16(a1, b1, acc[1][1].v);
    }

    const int half = lane >> 4, nl = lane & 15;
#pragma unroll
    for (int ni = 0; ni < 2; ++ni) {
        const int n = n0 + ni * 16 + nl;
        const int which = n / Dn;      // 0=q 1=k 2=v
        const int d  = n % Dn;
        const int h  = d / HDn;
        const int hd = d % HDn;
#pragma unroll
        for (int mi = 0; mi < 2; ++mi) {
#pragma unroll
            for (int i = 0; i < 8; ++i) {
                const int m  = m0 + mi * 16 + i + half * 8;
                const int bb = m / Ln, l = m % Ln;
                const _Float16 val = (_Float16)acc[mi][ni].f[i];
                const size_t bh = (size_t)(bb * Hn + h);
                if (which == 0)      Qb [(bh * Ln + l) * HDn + hd] = val;
                else if (which == 1) Kb [(bh * Ln + l) * HDn + hd] = val;
                else                 Vtb[(bh * HDn + hd) * Ln + l] = val;
            }
        }
    }
}

// ---------------------------------------------------------------------------
// 3) Flash attention, dual-path scores, causal, online softmax.
//    1 wave per (b,h,q-tile of 16). Wave-uniform trip count -> EXEC all 1s.
// ---------------------------------------------------------------------------
__global__ __launch_bounds__(128) void attn_kernel(
    const _Float16* __restrict__ Q, const _Float16* __restrict__ Kh,
    const _Float16* __restrict__ Vt, const _Float16* __restrict__ G,
    _Float16* __restrict__ Yb) {
    const int lane = threadIdx.x & 31;
    const int wave = threadIdx.x >> 5;
    const int job  = blockIdx.x * 4 + wave;   // 0..4095
    const int qt   = job & 127;
    const int bh   = job >> 7;                // 0..31
    const int b    = bh >> 4, h = bh & 15;
    const int qbase = qt * 16;
    const int half = lane >> 4, nl = lane & 15;

    __shared__ __align__(16) _Float16 Psh[4][16 * 32];
    _Float16* P = Psh[wave];

    // hoisted A fragments: q path and g path, HD=64 -> two K=32 frags each
    const size_t qk_base = ((size_t)bh * Ln + qbase) * HDn;
    const v16h aq0 = load_fragA(Q + qk_base,      HDn, lane);
    const v16h aq1 = load_fragA(Q + qk_base + 32, HDn, lane);
    const v16h ag0 = load_fragA(G + qk_base,      HDn, lane);
    const v16h ag1 = load_fragA(G + qk_base + 32, HDn, lane);

    float mrow[8], lrow[8];
    AccF y[4];
#pragma unroll
    for (int i = 0; i < 8; ++i) { mrow[i] = -__builtin_inff(); lrow[i] = 0.0f; }
#pragma unroll
    for (int t = 0; t < 4; ++t)
#pragma unroll
        for (int i = 0; i < 8; ++i) y[t].f[i] = 0.0f;

    const int kmax = qbase + 15;
    const int nblk = kmax / 32 + 1;           // uniform across the wave

    for (int j = 0; j < nblk; ++j) {
        const int kb = j * 32;
        float s[2][8], p[2][8], tmax[8];
#pragma unroll
        for (int i = 0; i < 8; ++i) tmax[i] = -__builtin_inff();

#pragma unroll
        for (int sub = 0; sub < 2; ++sub) {
            const int n0k = kb + sub * 16;
            const size_t kbase = ((size_t)bh * Ln + n0k) * HDn;
            const v16h bk0 = load_fragB(Kh + kbase,      HDn, lane);
            const v16h bk1 = load_fragB(Kh + kbase + 32, HDn, lane);
            const v16h bg0 = load_fragB(G  + kbase,      HDn, lane);
            const v16h bg1 = load_fragB(G  + kbase + 32, HDn, lane);
            AccF sx, sg;
#pragma unroll
            for (int i = 0; i < 8; ++i) { sx.f[i] = 0.0f; sg.f[i] = 0.0f; }
            sx.v = wmma16(aq0, bk0, sx.v);
            sx.v = wmma16(aq1, bk1, sx.v);
            sg.v = wmma16(ag0, bg0, sg.v);
            sg.v = wmma16(ag1, bg1, sg.v);

            const int kidx = n0k + nl;
#pragma unroll
            for (int i = 0; i < 8; ++i) {
                const int qidx = qbase + i + half * 8;
                const float sv = (kidx <= qidx)
                    ? SCALE * (1e-6f * sx.f[i] + sg.f[i])
                    : -__builtin_inff();
                s[sub][i] = sv;
                tmax[i] = fmaxf(tmax[i], sv);
            }
        }

        // row max over the 16-lane group (rows live per-VGPR across lanes)
#pragma unroll
        for (int i = 0; i < 8; ++i) {
            float v = tmax[i];
            v = fmaxf(v, __shfl_xor(v, 1));
            v = fmaxf(v, __shfl_xor(v, 2));
            v = fmaxf(v, __shfl_xor(v, 4));
            v = fmaxf(v, __shfl_xor(v, 8));
            tmax[i] = v;
        }

#pragma unroll
        for (int i = 0; i < 8; ++i) {
            const float mnew  = fmaxf(mrow[i], tmax[i]);
            const float alpha = __expf(mrow[i] - mnew);
            mrow[i] = mnew;
#pragma unroll
            for (int t = 0; t < 4; ++t) y[t].f[i] *= alpha;
            const float p0 = __expf(s[0][i] - mnew);
            const float p1 = __expf(s[1][i] - mnew);
            p[0][i] = p0; p[1][i] = p1;
            float rs = p0 + p1;
            rs += __shfl_xor(rs, 1);
            rs += __shfl_xor(rs, 2);
            rs += __shfl_xor(rs, 4);
            rs += __shfl_xor(rs, 8);
            lrow[i] = lrow[i] * alpha + rs;
        }

        // C-layout -> A-layout for P via LDS (row-major 16x32 f16, 1KB/wave)
#pragma unroll
        for (int i = 0; i < 8; ++i) {
            const int r = i + half * 8;
            P[r * 32 + nl]      = (_Float16)p[0][i];
            P[r * 32 + 16 + nl] = (_Float16)p[1][i];
        }
        asm volatile("s_wait_dscnt 0" ::: "memory");   // wave-local LDS RAW
        const v16h pf = load_fragA(P, 32, lane);

        if (j + 1 < nblk) {  // -> global_prefetch_b8 for next K block
            __builtin_prefetch(Kh + ((size_t)bh * Ln + kb + 32) * HDn, 0, 3);
        }

#pragma unroll
        for (int t = 0; t < 4; ++t) {
            const size_t vbase = ((size_t)bh * HDn + t * 16) * Ln + kb;
            const v16h vf = load_fragB(Vt + vbase, Ln, lane);
            y[t].v = wmma16(pf, vf, y[t].v);
        }
    }

    float inv[8];
#pragma unroll
    for (int i = 0; i < 8; ++i) inv[i] = 1.0f / lrow[i];
#pragma unroll
    for (int t = 0; t < 4; ++t) {
        const int col = h * HDn + t * 16 + nl;
#pragma unroll
        for (int i = 0; i < 8; ++i) {
            const int q = qbase + i + half * 8;
            Yb[((size_t)(b * Ln + q)) * Dn + col] = (_Float16)(y[t].f[i] * inv[i]);
        }
    }
}

// ---------------------------------------------------------------------------
// 4) Output projection: Yb[4096x1024] @ WoutT -> fp32 d_out
//    block = 4 waves; each wave: 32x32 C macro-tile (2x2 WMMA tiles)
// ---------------------------------------------------------------------------
__global__ __launch_bounds__(128) void out_gemm_kernel(
    const _Float16* __restrict__ A, const _Float16* __restrict__ Bt,
    float* __restrict__ out) {
    const int lane = threadIdx.x & 31;
    const int wave = threadIdx.x >> 5;
    const int m0   = blockIdx.y * 32;
    const int n0   = blockIdx.x * 128 + wave * 32;

    AccF acc[2][2];
#pragma unroll
    for (int mi = 0; mi < 2; ++mi)
#pragma unroll
        for (int ni = 0; ni < 2; ++ni)
#pragma unroll
            for (int i = 0; i < 8; ++i) acc[mi][ni].f[i] = 0.0f;

    const _Float16* arow = A  + (size_t)m0 * Dn;
    const _Float16* brow = Bt + (size_t)n0 * Dn;
    for (int k0 = 0; k0 < Dn; k0 += 32) {
        v16h a0 = load_fragA(arow + k0,                   Dn, lane);
        v16h a1 = load_fragA(arow + (size_t)16 * Dn + k0, Dn, lane);
        v16h b0 = load_fragB(brow + k0,                   Dn, lane);
        v16h b1 = load_fragB(brow + (size_t)16 * Dn + k0, Dn, lane);
        acc[0][0].v = wmma16(a0, b0, acc[0][0].v);
        acc[0][1].v = wmma16(a0, b1, acc[0][1].v);
        acc[1][0].v = wmma16(a1, b0, acc[1][0].v);
        acc[1][1].v = wmma16(a1, b1, acc[1][1].v);
    }

    const int half = lane >> 4, nl = lane & 15;
#pragma unroll
    for (int mi = 0; mi < 2; ++mi)
#pragma unroll
        for (int ni = 0; ni < 2; ++ni)
#pragma unroll
            for (int i = 0; i < 8; ++i) {
                const int m = m0 + mi * 16 + i + half * 8;
                out[(size_t)m * Dn + n0 + ni * 16 + nl] = acc[mi][ni].f[i];
            }
}

// ---------------------------------------------------------------------------
extern "C" void kernel_launch(void* const* d_in, const int* in_sizes, int n_in,
                              void* d_out, int out_size, void* d_ws, size_t ws_size,
                              hipStream_t stream) {
    const float* x    = (const float*)d_in[0];
    const float* g    = (const float*)d_in[1];
    const float* wqkv = (const float*)d_in[2];
    const float* wout = (const float*)d_in[3];
    float* out = (float*)d_out;

    _Float16* ws    = (_Float16*)d_ws;      // needs ~56 MB of scratch
    _Float16* Xb    = ws + XB_OFF;
    _Float16* WqkvT = ws + WQKVT_OFF;
    _Float16* WoutT = ws + WOUTT_OFF;
    _Float16* Gb    = ws + GB_OFF;
    _Float16* Qb    = ws + QB_OFF;
    _Float16* Kb    = ws + KB_OFF;
    _Float16* Vtb   = ws + VTB_OFF;
    _Float16* Yb    = ws + YB_OFF;

    // 1) precision conversion + layout transforms
    convert_kernel<<<(M1 * Dn + 255) / 256, 256, 0, stream>>>(
        x, g, wqkv, wout, Xb, Gb, WqkvT, WoutT);

    // 2) QKV projection (WMMA f16, 32x32 per wave)
    qkv_gemm_kernel<<<dim3(N1 / 128, M1 / 32), 128, 0, stream>>>(
        Xb, WqkvT, Qb, Kb, Vtb);

    // 3) causal dual-path flash attention (WMMA f16), 4096 wave-jobs / 4 waves per block
    attn_kernel<<<(Bn * Hn * (Ln / 16)) / 4, 128, 0, stream>>>(
        Qb, Kb, Vtb, Gb, Yb);

    // 4) output projection (WMMA f16 -> fp32, 32x32 per wave)
    out_gemm_kernel<<<dim3(Dn / 128, M1 / 32), 128, 0, stream>>>(
        Yb, WoutT, out);
}
